// MCANet_61357902790897
// MI455X (gfx1250) — matrix operand order repaired and checked by reading, not compiled, not run on hardware
//
#include <hip/hip_runtime.h>
#include <hip/hip_bf16.h>

#define G_   4096
#define P_   21
#define DEG_ 16
#define CH_  6
#define N_   (G_*P_)
#define E_   (N_*DEG_)

typedef __attribute__((ext_vector_type(16))) _Float16 v16h;
typedef __attribute__((ext_vector_type(8)))  float    v8f;

union AV { uint4 u[2]; v16h v; _Float16 h[16]; };

__device__ __forceinline__ float leaky(float x) { return x > 0.f ? x : 0.01f * x; }

#if __has_builtin(__builtin_amdgcn_tanhf)
__device__ __forceinline__ float tanhx(float x) { return __builtin_amdgcn_tanhf(x); }
#else
__device__ __forceinline__ float tanhx(float x) {
  x = fminf(fmaxf(x, -15.f), 15.f);
  float t = __expf(2.f * x);
  return 1.f - 2.f * __builtin_amdgcn_rcpf(t + 1.f);
}
#endif
// sigmoid(x) = 0.5*tanh(x/2) + 0.5  -> one TRANS + one FMA
__device__ __forceinline__ float sigf(float x) { return fmaf(tanhx(0.5f * x), 0.5f, 0.5f); }

// ---------------- prep: fold edge-MLP to affine, pre-swizzle WMMA B tiles ---------------
__global__ void prep_kernel(const float* W_wfc, const float* b_wfc, const float* w_wfc1,
                            const float* b_wfc1, const float* b_ih2, const float* b_hh2,
                            const float* W_ih2, const float* W_hh2,
                            const float* W_self2, const float* W_neigh2,
                            const float* bn_gamma, const float* bn_beta,
                            const float* bn_mean, const float* bn_var,
                            float* scal, float* bias2, float* bnscale, float* bnshift,
                            _Float16* Bih, _Float16* Bhh, _Float16* Bself, _Float16* Bneigh) {
  int tid = threadIdx.x;
  if (tid == 0) {
    float c0 = 0.f, c1 = 0.f, c2 = b_wfc1[0];
    for (int k = 0; k < 100; ++k) {
      float w = w_wfc1[k];
      c0 += W_wfc[k * 2 + 0] * w;
      c1 += W_wfc[k * 2 + 1] * w;
      c2 += b_wfc[k] * w;
    }
    scal[0] = c0; scal[1] = c1; scal[2] = c2;
  }
  if (tid < 128) bias2[tid] = b_ih2[tid] + b_hh2[tid];
  if (tid < 20) {
    float sc = bn_gamma[tid] * rsqrtf(bn_var[tid] + 1e-5f);
    bnscale[tid] = sc;
    bnshift[tid] = bn_beta[tid] - bn_mean[tid] * sc;
  }
  // B operand layout: entry (T,l,h): lane l supplies column n=l&15 (gate 16T+n),
  // halves = K = 16*(h>>3) + 8*(l>>4) + (h&7)  (table K order per lane-half).
  for (int i = tid; i < 8 * 32 * 16; i += blockDim.x) {
    int T = i >> 9, l = (i >> 4) & 31, h = i & 15;
    int K = ((h >> 3) << 4) + ((l >> 4) << 3) + (h & 7);
    int g = (T << 4) + (l & 15);
    Bih[i] = (_Float16)W_ih2[g * 32 + K];
    Bhh[i] = (_Float16)W_hh2[g * 32 + K];
  }
  for (int i = tid; i < 2 * 32 * 16; i += blockDim.x) {
    int T = i >> 9, l = (i >> 4) & 31, h = i & 15;
    int K = ((h >> 3) << 4) + ((l >> 4) << 3) + (h & 7);
    int g = (T << 4) + (l & 15);
    Bself[i]  = (g < 20) ? (_Float16)W_self2[g * 32 + K]  : (_Float16)0.f;
    Bneigh[i] = (g < 20) ? (_Float16)W_neigh2[g * 32 + K] : (_Float16)0.f;
  }
}

// ---------------- per-edge weight ----------------
__global__ void ew_kernel(const float* dis, const float* spec, const float* scal, float* ew) {
  int e = blockIdx.x * blockDim.x + threadIdx.x;
  if (e < E_) ew[e] = scal[0] * dis[e] + scal[1] * spec[e] + scal[2];
}

// ---------------- channel attention -> feat [N,6] ----------------
__global__ void feat_kernel(const float* x, const float* W_hfc, const float* b_hfc,
                            const float* w_attn2, float* feat) {
  int n = blockIdx.x * blockDim.x + threadIdx.x;
  if (n >= N_) return;
  float h[CH_], z[CH_];
#pragma unroll
  for (int c = 0; c < CH_; ++c) h[c] = x[n * CH_ + c];
#pragma unroll
  for (int i = 0; i < CH_; ++i) {
    float s = b_hfc[i];
#pragma unroll
    for (int c = 0; c < CH_; ++c) s += h[c] * W_hfc[i * CH_ + c];
    z[i] = s;
  }
  float s0 = 0.f, s1 = 0.f;
#pragma unroll
  for (int j = 0; j < CH_; ++j) { s0 += z[j] * w_attn2[j]; s1 += w_attn2[CH_ + j]; }
  float a[CH_], mx = -1e30f;
#pragma unroll
  for (int i = 0; i < CH_; ++i) { a[i] = leaky(s0 + z[i] * s1); mx = fmaxf(mx, a[i]); }
  float sum = 0.f;
#pragma unroll
  for (int i = 0; i < CH_; ++i) { a[i] = __expf(a[i] - mx); sum += a[i]; }
  float inv = __builtin_amdgcn_rcpf(sum);
#pragma unroll
  for (int i = 0; i < CH_; ++i) feat[n * CH_ + i] = a[i] * inv * z[i];
}

// ---------------- SAGE-1 (LSTM over 16 neighbors, D=6, scalar VALU) ----------------
__global__ void lstm1_kernel(const float* feat, const float* ewA, const int* src_local,
                             const float* W_ih, const float* W_hh, const float* b_ih,
                             const float* b_hh, const float* W_self, const float* W_neigh,
                             const float* b_sage, _Float16* h1out) {
  __shared__ float sWih[144], sWhh[144], sB[24], sWs[192], sWn[192], sBs[32];
  int tid = threadIdx.x;
  for (int i = tid; i < 144; i += blockDim.x) { sWih[i] = W_ih[i]; sWhh[i] = W_hh[i]; }
  for (int i = tid; i < 24; i += blockDim.x) sB[i] = b_ih[i] + b_hh[i];
  for (int i = tid; i < 192; i += blockDim.x) { sWs[i] = W_self[i]; sWn[i] = W_neigh[i]; }
  for (int i = tid; i < 32; i += blockDim.x) sBs[i] = b_sage[i];
  __syncthreads();
  int n = blockIdx.x * blockDim.x + tid;
  if (n >= N_) return;
  int gbase = (n / P_) * P_;
  float fown[CH_];
#pragma unroll
  for (int c = 0; c < CH_; ++c) fown[c] = feat[n * CH_ + c];
  float h[CH_] = {0, 0, 0, 0, 0, 0}, cst[CH_] = {0, 0, 0, 0, 0, 0};
  for (int t = 0; t < DEG_; ++t) {
    int e = n * DEG_ + t;
    int s = gbase + src_local[e];
    float w = ewA[e];
    float m[CH_];
#pragma unroll
    for (int c = 0; c < CH_; ++c) m[c] = feat[s * CH_ + c] * w;
    float g[24];
#pragma unroll
    for (int k = 0; k < 24; ++k) {
      float acc = sB[k];
#pragma unroll
      for (int c = 0; c < CH_; ++c) acc += m[c] * sWih[k * CH_ + c] + h[c] * sWhh[k * CH_ + c];
      g[k] = acc;
    }
#pragma unroll
    for (int j = 0; j < CH_; ++j) {
      float iv = sigf(g[j]), fv = sigf(g[CH_ + j]);
      float gv = tanhx(g[2 * CH_ + j]), ov = sigf(g[3 * CH_ + j]);
      cst[j] = fv * cst[j] + iv * gv;
      h[j] = ov * tanhx(cst[j]);
    }
  }
#pragma unroll
  for (int k = 0; k < 32; ++k) {
    float v = sBs[k];
#pragma unroll
    for (int c = 0; c < CH_; ++c) v += fown[c] * sWs[k * CH_ + c] + h[c] * sWn[k * CH_ + c];
    h1out[n * 32 + k] = (_Float16)leaky(v);
  }
}

// ---------------- SAGE-2: WMMA LSTM (16 nodes/wave, K=32, 128 gates) ----------------
#define WAVES 4
__global__ void __launch_bounds__(WAVES * 32)
lstm2_kernel(const _Float16* h1, const float* ewA, const int* src_local,
             const float* bias2, const _Float16* Bihp, const _Float16* Bhhp,
             const _Float16* Bselfp, const _Float16* Bneighp,
             const float* b_sage2, const float* bnscale, const float* bnshift, float* h2) {
  __shared__ _Float16 ldsH[WAVES * 16 * 32];
  int lane = threadIdx.x & 31;
  int wv = threadIdx.x >> 5;
  int group = blockIdx.x * WAVES + wv;
  int r = lane & 15;       // matrix row (A) / column (B,C,D) owned by this lane
  int kh = lane >> 4;      // lane-half selects K octets / row-high-8
  int base = group * 16;
  int nodeR = base + r;
  int gbase = (nodeR / P_) * P_;
  _Float16* lh = ldsH + wv * 512;

  v16h Bih[8], Bhh[8];
#pragma unroll
  for (int T = 0; T < 8; ++T) {
    Bih[T] = *(const v16h*)(Bihp + (T * 32 + lane) * 16);
    Bhh[T] = *(const v16h*)(Bhhp + (T * 32 + lane) * 16);
  }
  float bg[8];
#pragma unroll
  for (int T = 0; T < 8; ++T) bg[T] = bias2[T * 16 + r];

  v8f hS[2], cS[2];
#pragma unroll
  for (int s = 0; s < 2; ++s)
#pragma unroll
    for (int rr = 0; rr < 8; ++rr) { hS[s][rr] = 0.f; cS[s][rr] = 0.f; }

  for (int t = 0; t < DEG_; ++t) {
    // transpose h (C/D layout -> A layout) through wave-private LDS
#pragma unroll
    for (int s = 0; s < 2; ++s)
#pragma unroll
      for (int rr = 0; rr < 8; ++rr)
        lh[(rr + kh * 8) * 32 + s * 16 + r] = (_Float16)hS[s][rr];
    asm volatile("s_wait_dscnt 0x0" ::: "memory");
    AV ah;
    ah.u[0] = *(const uint4*)(lh + r * 32 + kh * 8);
    ah.u[1] = *(const uint4*)(lh + r * 32 + 16 + kh * 8);

    // gather neighbor row, scale by edge weight (packed f16) -> A operand
    int e = nodeR * DEG_ + t;
    int sidx = gbase + src_local[e];
    _Float16 wh = (_Float16)ewA[e];
    AV am;
    am.u[0] = *(const uint4*)(h1 + sidx * 32 + kh * 8);
    am.u[1] = *(const uint4*)(h1 + sidx * 32 + 16 + kh * 8);
    am.v = am.v * wh;   // v_pk_mul_f16 x8, no conversions

    v8f gA[8];
#pragma unroll
    for (int T = 0; T < 8; ++T)
#pragma unroll
      for (int rr = 0; rr < 8; ++rr) gA[T][rr] = bg[T];
#pragma unroll
    for (int T = 0; T < 8; ++T) {
      gA[T] = __builtin_amdgcn_wmma_f32_16x16x32_f16(false, am.v, false, Bih[T],
                                                     (short)0, gA[T], false, false);
      gA[T] = __builtin_amdgcn_wmma_f32_16x16x32_f16(false, ah.v, false, Bhh[T],
                                                     (short)0, gA[T], false, false);
    }
    // gates (i,f,g,o) each 32 wide -> tiles {0,1},{2,3},{4,5},{6,7}
#pragma unroll
    for (int s = 0; s < 2; ++s)
#pragma unroll
      for (int rr = 0; rr < 8; ++rr) {
        float iv = sigf(gA[s][rr]);
        float fv = sigf(gA[2 + s][rr]);
        float gv = tanhx(gA[4 + s][rr]);
        float ov = sigf(gA[6 + s][rr]);
        float cv = fv * cS[s][rr] + iv * gv;
        cS[s][rr] = cv;
        hS[s][rr] = ov * tanhx(cv);
      }
  }

  // final hn -> A layout
#pragma unroll
  for (int s = 0; s < 2; ++s)
#pragma unroll
    for (int rr = 0; rr < 8; ++rr)
      lh[(rr + kh * 8) * 32 + s * 16 + r] = (_Float16)hS[s][rr];
  asm volatile("s_wait_dscnt 0x0" ::: "memory");
  AV ahn, aself;
  ahn.u[0] = *(const uint4*)(lh + r * 32 + kh * 8);
  ahn.u[1] = *(const uint4*)(lh + r * 32 + 16 + kh * 8);
  aself.u[0] = *(const uint4*)(h1 + nodeR * 32 + kh * 8);
  aself.u[1] = *(const uint4*)(h1 + nodeR * 32 + 16 + kh * 8);

  v8f rA[2];
#pragma unroll
  for (int T = 0; T < 2; ++T) {
    int g = T * 16 + r;
    float bs = (g < 20) ? b_sage2[g] : 0.f;
#pragma unroll
    for (int rr = 0; rr < 8; ++rr) rA[T][rr] = bs;
    v16h Bs = *(const v16h*)(Bselfp + (T * 32 + lane) * 16);
    v16h Bn = *(const v16h*)(Bneighp + (T * 32 + lane) * 16);
    rA[T] = __builtin_amdgcn_wmma_f32_16x16x32_f16(false, aself.v, false, Bs,
                                                   (short)0, rA[T], false, false);
    rA[T] = __builtin_amdgcn_wmma_f32_16x16x32_f16(false, ahn.v, false, Bn,
                                                   (short)0, rA[T], false, false);
  }
  // BN + leaky, scatter to h2[N][20]
#pragma unroll
  for (int T = 0; T < 2; ++T) {
    int g = T * 16 + r;
    if (g < 20) {
      float sc = bnscale[g], sh = bnshift[g];
#pragma unroll
      for (int rr = 0; rr < 8; ++rr) {
        int M = rr + kh * 8;
        h2[(base + M) * 20 + g] = leaky(rA[T][rr] * sc + sh);
      }
    }
  }
}

// ---------------- global attention pooling + FC head ----------------
__global__ void pool_kernel(const float* h2, const float* w_gate, const float* b_gate,
                            const float* W_fc1, const float* b_fc1, const float* W_fc2,
                            const float* b_fc2, float* out) {
  int g = blockIdx.x * blockDim.x + threadIdx.x;
  if (g >= G_) return;
  const float* hb = h2 + (size_t)g * P_ * 20;
  float gate[P_], mx = -1e30f;
  for (int p = 0; p < P_; ++p) {
    float s = b_gate[0];
#pragma unroll
    for (int d = 0; d < 20; ++d) s += hb[p * 20 + d] * w_gate[d];
    gate[p] = s;
    mx = fmaxf(mx, s);
  }
  float sum = 0.f;
  for (int p = 0; p < P_; ++p) { gate[p] = __expf(gate[p] - mx); sum += gate[p]; }
  float inv = __builtin_amdgcn_rcpf(sum);
  float pooled[20];
#pragma unroll
  for (int d = 0; d < 20; ++d) pooled[d] = 0.f;
  for (int p = 0; p < P_; ++p) {
    float w = gate[p] * inv;
#pragma unroll
    for (int d = 0; d < 20; ++d) pooled[d] += w * hb[p * 20 + d];
  }
  float f1[10];
#pragma unroll
  for (int k = 0; k < 10; ++k) {
    float v = b_fc1[k];
#pragma unroll
    for (int d = 0; d < 20; ++d) v += pooled[d] * W_fc1[k * 20 + d];
    f1[k] = leaky(v);
  }
#pragma unroll
  for (int j = 0; j < 2; ++j) {
    float v = b_fc2[j];
#pragma unroll
    for (int k = 0; k < 10; ++k) v += f1[k] * W_fc2[j * 10 + k];
    out[g * 2 + j] = v;
  }
}

// ---------------- workspace layout (bytes) ----------------
#define WS_SCAL    0u
#define WS_BIAS2   256u
#define WS_BNSC    768u
#define WS_BNSH    896u
#define WS_BIH     1024u
#define WS_BHH     (WS_BIH + 8192u)
#define WS_BSELF   (WS_BHH + 8192u)
#define WS_BNEIGH  (WS_BSELF + 2048u)
#define WS_EW      (WS_BNEIGH + 2048u)
#define WS_FEAT    (WS_EW + (unsigned)E_ * 4u)
#define WS_H1      (WS_FEAT + (unsigned)N_ * CH_ * 4u)
#define WS_H2      (WS_H1 + (unsigned)N_ * 64u)

extern "C" void kernel_launch(void* const* d_in, const int* in_sizes, int n_in,
                              void* d_out, int out_size, void* d_ws, size_t ws_size,
                              hipStream_t stream) {
  const float* x        = (const float*)d_in[0];
  const float* dis      = (const float*)d_in[1];
  const float* spec     = (const float*)d_in[2];
  const float* W_hfc    = (const float*)d_in[3];
  const float* b_hfc    = (const float*)d_in[4];
  const float* w_attn2  = (const float*)d_in[5];
  const float* W_wfc    = (const float*)d_in[6];
  const float* b_wfc    = (const float*)d_in[7];
  const float* w_wfc1   = (const float*)d_in[8];
  const float* b_wfc1   = (const float*)d_in[9];
  const float* l1_Wih   = (const float*)d_in[10];
  const float* l1_Whh   = (const float*)d_in[11];
  const float* l1_bih   = (const float*)d_in[12];
  const float* l1_bhh   = (const float*)d_in[13];
  const float* W_self1  = (const float*)d_in[14];
  const float* W_neigh1 = (const float*)d_in[15];
  const float* b_sage1  = (const float*)d_in[16];
  const float* l2_Wih   = (const float*)d_in[17];
  const float* l2_Whh   = (const float*)d_in[18];
  const float* l2_bih   = (const float*)d_in[19];
  const float* l2_bhh   = (const float*)d_in[20];
  const float* W_self2  = (const float*)d_in[21];
  const float* W_neigh2 = (const float*)d_in[22];
  const float* b_sage2  = (const float*)d_in[23];
  const float* bn_gamma = (const float*)d_in[24];
  const float* bn_beta  = (const float*)d_in[25];
  const float* bn_mean  = (const float*)d_in[26];
  const float* bn_var   = (const float*)d_in[27];
  const float* w_gate   = (const float*)d_in[28];
  const float* b_gate   = (const float*)d_in[29];
  const float* W_fc1    = (const float*)d_in[30];
  const float* b_fc1    = (const float*)d_in[31];
  const float* W_fc2    = (const float*)d_in[32];
  const float* b_fc2    = (const float*)d_in[33];
  const int*   src_loc  = (const int*)d_in[34];
  float* out = (float*)d_out;

  char* ws = (char*)d_ws;
  float*     scal    = (float*)(ws + WS_SCAL);
  float*     bias2   = (float*)(ws + WS_BIAS2);
  float*     bnscale = (float*)(ws + WS_BNSC);
  float*     bnshift = (float*)(ws + WS_BNSH);
  _Float16*  Bih     = (_Float16*)(ws + WS_BIH);
  _Float16*  Bhh     = (_Float16*)(ws + WS_BHH);
  _Float16*  Bself   = (_Float16*)(ws + WS_BSELF);
  _Float16*  Bneigh  = (_Float16*)(ws + WS_BNEIGH);
  float*     ew      = (float*)(ws + WS_EW);
  float*     feat    = (float*)(ws + WS_FEAT);
  _Float16*  h1f16   = (_Float16*)(ws + WS_H1);
  float*     h2      = (float*)(ws + WS_H2);

  prep_kernel<<<1, 256, 0, stream>>>(W_wfc, b_wfc, w_wfc1, b_wfc1, l2_bih, l2_bhh,
                                     l2_Wih, l2_Whh, W_self2, W_neigh2,
                                     bn_gamma, bn_beta, bn_mean, bn_var,
                                     scal, bias2, bnscale, bnshift, Bih, Bhh, Bself, Bneigh);
  ew_kernel<<<(E_ + 255) / 256, 256, 0, stream>>>(dis, spec, scal, ew);
  feat_kernel<<<(N_ + 255) / 256, 256, 0, stream>>>(x, W_hfc, b_hfc, w_attn2, feat);
  lstm1_kernel<<<(N_ + 255) / 256, 256, 0, stream>>>(feat, ew, src_loc, l1_Wih, l1_Whh,
                                                     l1_bih, l1_bhh, W_self1, W_neigh1,
                                                     b_sage1, h1f16);
  lstm2_kernel<<<(N_ / 16) / WAVES, WAVES * 32, 0, stream>>>(h1f16, ew, src_loc, bias2,
                                                             Bih, Bhh, Bself, Bneigh,
                                                             b_sage2, bnscale, bnshift, h2);
  pool_kernel<<<(G_ + 255) / 256, 256, 0, stream>>>(h2, w_gate, b_gate, W_fc1, b_fc1,
                                                    W_fc2, b_fc2, out);
}
